// LinearCrossAttention_2044404433519
// MI455X (gfx1250) — matrix operand order repaired and checked by reading, not compiled, not run on hardware
//
#include <hip/hip_runtime.h>
#include <hip/hip_bf16.h>

// Linear cross-attention for MI455X (gfx1250, wave32, WMMA).
// All GEMMs use V_WMMA_F32_16X16X32_BF16 (bf16 A/B, f32 accumulate).
// Projection GEMMs stage the shared W tile through LDS (double-buffered),
// so the B operand is loaded from global once per block instead of once
// per wave; inner loop is {ds_load_b128, global_load_b128, v_wmma}.

#define T_SEQ  4096
#define BATCH  4
#define DMODEL 1024
#define NHEAD  16
#define HDIM   64
#define LDSTRIDE 40   // halfs per LDS row: 80B = 16B-aligned + conflict-free

typedef __attribute__((ext_vector_type(16))) __bf16 v16bf;
typedef __attribute__((ext_vector_type(8)))  __bf16 v8bf;
typedef __attribute__((ext_vector_type(8)))  float  v8f;
typedef __attribute__((ext_vector_type(4)))  float  v4f;

static __device__ __forceinline__ __bf16 f2bf(float f) {
  union { float f; unsigned u; } a; a.f = f;
  unsigned r = a.u + 0x7FFFu + ((a.u >> 16) & 1u);   // round-to-nearest-even
  unsigned short h = (unsigned short)(r >> 16);
  __bf16 b; __builtin_memcpy(&b, &h, 2);
  return b;
}

static __device__ __forceinline__ v8f wmma_bf16(v16bf a, v16bf b, v8f c) {
  // v_wmma_f32_16x16x32_bf16: (neg_a, A, neg_b, B, c_mod, C, reuse_a, reuse_b)
  return __builtin_amdgcn_wmma_f32_16x16x32_bf16(false, a, false, b, (short)0, c,
                                                 false, false);
}

static __device__ __forceinline__ v16bf cat16(v8bf a0, v8bf a1) {
  return __builtin_shufflevector(a0, a1, 0, 1, 2, 3, 4, 5, 6, 7,
                                         8, 9, 10, 11, 12, 13, 14, 15);
}

// branchless elu(x)+1 : x>0 ? x+1 : exp(x)
static __device__ __forceinline__ float elu1(float x) {
  float e = __expf(fminf(x, 0.f));
  return x > 0.f ? x + 1.f : e;
}

// ---------------------------------------------------------------- convert ---
__global__ void convert_bf16x8_kernel(const float* __restrict__ src,
                                      __bf16* __restrict__ dst, int n8) {
  int i = blockIdx.x * blockDim.x + threadIdx.x;
  const int stride = gridDim.x * blockDim.x;
  for (; i < n8; i += stride) {
    v4f f0 = *(const v4f*)(src + (size_t)i * 8);
    v4f f1 = *(const v4f*)(src + (size_t)i * 8 + 4);
    v8bf pk;
#pragma unroll
    for (int k = 0; k < 4; ++k) pk[k]     = f2bf(f0[k]);
#pragma unroll
    for (int k = 0; k < 4; ++k) pk[4 + k] = f2bf(f1[k]);
    *(v8bf*)(dst + (size_t)i * 8) = pk;
  }
}

// -------------------------------------------------------- QKV projections ---
// out[n,j] = act( X[n,:] . W[j,:] + bias[j] )
// Block tile 128x64, wave tile 16x64 (4 accumulators). W slab via LDS.
// mode 0: elu+1, store q' as [b,h,t,d]   (A-major for attn stage)
// mode 1: elu+1, store k'^T as [b,h,d,T] (A-major for context stage)
// mode 2: none,  store v^T  as [b,h,d,T] (B-major for context stage)
__global__ void qkv_proj_kernel(const __bf16* __restrict__ Xbf,
                                const __bf16* __restrict__ Wbf,
                                const float* __restrict__ bias,
                                __bf16* __restrict__ outp, int mode) {
  __shared__ __attribute__((aligned(16))) __bf16 wbuf[2][64 * LDSTRIDE];

  const int lane  = threadIdx.x & 31;
  const int wave  = threadIdx.x >> 5;
  const int hi    = lane >> 4;          // lane half selects K chunk
  const int ml    = lane & 15;          // A row / B col within tile
  const int tileT = blockIdx.x * 128 + wave * 16;
  const int n0    = blockIdx.y * 64;
  const int b     = blockIdx.z;

  const int t = tileT + ml;
  const __bf16* xrow = Xbf + ((size_t)t * BATCH + b) * DMODEL;
  // staging role: thread -> (row j of W slab, 8-half chunk kc)
  const int sj = threadIdx.x >> 2;
  const int sk = (threadIdx.x & 3) * 8;
  const __bf16* wsrc = Wbf + (size_t)(n0 + sj) * DMODEL + sk;

  // stage k0 = 0 slab into buffer 0
  *(v8bf*)(&wbuf[0][sj * LDSTRIDE + sk]) = *(const v8bf*)(wsrc);
  __syncthreads();

  v8f acc[4] = {};
  for (int k0 = 0; k0 < DMODEL; k0 += 32) {
    const int cur = (k0 >> 5) & 1;
    if (k0 + 32 < DMODEL)   // prefetch next slab into other buffer
      *(v8bf*)(&wbuf[cur ^ 1][sj * LDSTRIDE + sk]) =
          *(const v8bf*)(wsrc + k0 + 32);

    v16bf a = cat16(*(const v8bf*)(xrow + k0 + hi * 16),
                    *(const v8bf*)(xrow + k0 + hi * 16 + 8));
#pragma unroll
    for (int nt = 0; nt < 4; ++nt) {
      const __bf16* bsrc = &wbuf[cur][(nt * 16 + ml) * LDSTRIDE + hi * 16];
      v16bf bm = cat16(*(const v8bf*)(bsrc), *(const v8bf*)(bsrc + 8));
      acc[nt] = wmma_bf16(a, bm, acc[nt]);
    }
    __syncthreads();
  }

#pragma unroll
  for (int nt = 0; nt < 4; ++nt) {
    const int j  = n0 + nt * 16 + ml;
    const float bj = bias[j];
    const int h  = j >> 6;
    const int dh = j & 63;
    if (mode == 0) {
#pragma unroll
      for (int r = 0; r < 8; ++r) {
        const float vv = elu1(acc[nt][r] + bj);
        const int tr = tileT + hi * 8 + r;
        outp[(((size_t)b * NHEAD + h) * T_SEQ + tr) * HDIM + dh] = f2bf(vv);
      }
    } else {
      v8bf pk;
#pragma unroll
      for (int r = 0; r < 8; ++r) {
        float vv = acc[nt][r] + bj;
        if (mode == 1) vv = elu1(vv);
        pk[r] = f2bf(vv);
      }
      // transposed layout: consecutive r are consecutive t -> one 16B store
      __bf16* dst = outp + (((size_t)b * NHEAD + h) * HDIM + dh) * T_SEQ
                         + tileT + hi * 8;
      *(v8bf*)dst = pk;
    }
  }
}

// --------------------------------------------- context = k'^T @ v  (64x64) --
// 8 waves: 4 m-tiles x 2 K-halves; K-halves reduced through LDS.
__global__ void context_kernel(const __bf16* __restrict__ kpT,
                               const __bf16* __restrict__ vT,
                               __bf16* __restrict__ ctxT) {
  __shared__ float red[4 * 32 * 32];            // 16 KB: [mtile][lane][4x8 f32]
  const int lane  = threadIdx.x & 31;
  const int wave  = threadIdx.x >> 5;
  const int hi    = lane >> 4;
  const int ml    = lane & 15;
  const int bh    = blockIdx.x;
  const int m0    = (wave & 3) * 16;            // d-row tile
  const int khalf = wave >> 2;                  // K split: [0,2048) / [2048,4096)

  const __bf16* Abase = kpT + ((size_t)bh * HDIM + m0 + ml) * T_SEQ;
  const __bf16* Bcol  = vT  + ((size_t)bh * HDIM + ml) * T_SEQ;

  v8f acc[4] = {};
  const int kbeg = khalf * (T_SEQ / 2);
  const int kend = kbeg + (T_SEQ / 2);
  for (int k0 = kbeg; k0 < kend; k0 += 32) {
    v16bf a = cat16(*(const v8bf*)(Abase + k0 + hi * 16),
                    *(const v8bf*)(Abase + k0 + hi * 16 + 8));
#pragma unroll
    for (int nt = 0; nt < 4; ++nt) {
      v16bf bm = *(const v16bf*)(Bcol + (size_t)nt * 16 * T_SEQ + k0 + hi * 16);
      acc[nt] = wmma_bf16(a, bm, acc[nt]);
    }
  }

  float* slot = red + ((wave & 3) * 32 + lane) * 32;
  if (khalf == 1) {
#pragma unroll
    for (int nt = 0; nt < 4; ++nt)
#pragma unroll
      for (int r = 0; r < 8; ++r) slot[nt * 8 + r] = acc[nt][r];
  }
  __syncthreads();
  if (khalf == 0) {
#pragma unroll
    for (int nt = 0; nt < 4; ++nt) {
      v8bf pk;
#pragma unroll
      for (int r = 0; r < 8; ++r) pk[r] = f2bf(acc[nt][r] + slot[nt * 8 + r]);
      // store transposed (ctxT[e][d]) so attn-stage B-fragments are contiguous
      __bf16* dst = ctxT + ((size_t)bh * HDIM + nt * 16 + ml) * HDIM
                         + m0 + hi * 8;
      *(v8bf*)dst = pk;
    }
  }
}

// --------------------------------------------------- attn = q' @ context ----
__global__ void attn_kernel(const __bf16* __restrict__ qp,
                            const __bf16* __restrict__ ctxT,
                            __bf16* __restrict__ attn) {
  const int lane  = threadIdx.x & 31;
  const int wave  = threadIdx.x >> 5;
  const int hi    = lane >> 4;
  const int ml    = lane & 15;
  const int tileT = blockIdx.x * 128 + wave * 16;
  const int bh    = blockIdx.y;
  const int b     = bh >> 4;
  const int h     = bh & 15;

  const __bf16* Abase = qp + (((size_t)bh * T_SEQ) + tileT + ml) * HDIM;
  // A fragments for K=0..31 and K=32..63, loaded once, reused for 4 n-tiles
  v16bf ak0 = cat16(*(const v8bf*)(Abase + hi * 16),
                    *(const v8bf*)(Abase + hi * 16 + 8));
  v16bf ak1 = cat16(*(const v8bf*)(Abase + 32 + hi * 16),
                    *(const v8bf*)(Abase + 32 + hi * 16 + 8));

#pragma unroll
  for (int nt = 0; nt < 4; ++nt) {
    const __bf16* Bbase = ctxT + ((size_t)bh * HDIM + nt * 16 + ml) * HDIM;
    v16bf bm0 = *(const v16bf*)(Bbase + hi * 16);
    v16bf bm1 = *(const v16bf*)(Bbase + 32 + hi * 16);
    v8f acc = {};
    acc = wmma_bf16(ak0, bm0, acc);
    acc = wmma_bf16(ak1, bm1, acc);
    const int col = h * 64 + nt * 16 + ml;
#pragma unroll
    for (int r = 0; r < 8; ++r) {
      const int tr = tileT + hi * 8 + r;
      attn[((size_t)b * T_SEQ + tr) * DMODEL + col] = f2bf(acc[r]);
    }
  }
}

// --------------------------------------------------- out = attn @ Wo^T + bo -
__global__ void out_proj_kernel(const __bf16* __restrict__ attn,
                                const __bf16* __restrict__ WoBf,
                                const float* __restrict__ bo,
                                float* __restrict__ out) {
  __shared__ __attribute__((aligned(16))) __bf16 wbuf[2][64 * LDSTRIDE];

  const int lane  = threadIdx.x & 31;
  const int wave  = threadIdx.x >> 5;
  const int hi    = lane >> 4;
  const int ml    = lane & 15;
  const int tileT = blockIdx.x * 128 + wave * 16;
  const int n0    = blockIdx.y * 64;
  const int b     = blockIdx.z;

  const __bf16* Arow = attn + ((size_t)b * T_SEQ + tileT + ml) * DMODEL;
  const int sj = threadIdx.x >> 2;
  const int sk = (threadIdx.x & 3) * 8;
  const __bf16* wsrc = WoBf + (size_t)(n0 + sj) * DMODEL + sk;

  *(v8bf*)(&wbuf[0][sj * LDSTRIDE + sk]) = *(const v8bf*)(wsrc);
  __syncthreads();

  v8f acc[4] = {};
  for (int k0 = 0; k0 < DMODEL; k0 += 32) {
    const int cur = (k0 >> 5) & 1;
    if (k0 + 32 < DMODEL)
      *(v8bf*)(&wbuf[cur ^ 1][sj * LDSTRIDE + sk]) =
          *(const v8bf*)(wsrc + k0 + 32);

    v16bf a = cat16(*(const v8bf*)(Arow + k0 + hi * 16),
                    *(const v8bf*)(Arow + k0 + hi * 16 + 8));
#pragma unroll
    for (int nt = 0; nt < 4; ++nt) {
      const __bf16* bsrc = &wbuf[cur][(nt * 16 + ml) * LDSTRIDE + hi * 16];
      v16bf bm = cat16(*(const v8bf*)(bsrc), *(const v8bf*)(bsrc + 8));
      acc[nt] = wmma_bf16(a, bm, acc[nt]);
    }
    __syncthreads();
  }

#pragma unroll
  for (int nt = 0; nt < 4; ++nt) {
    const int j  = n0 + nt * 16 + ml;
    const float bj = bo[j];
#pragma unroll
    for (int r = 0; r < 8; ++r) {
      const int tr = tileT + hi * 8 + r;
      out[((size_t)tr * BATCH + b) * DMODEL + j] = acc[nt][r] + bj;  // [T,B,D]
    }
  }
}

// ---------------------------------------------------------------- launcher --
extern "C" void kernel_launch(void* const* d_in, const int* in_sizes, int n_in,
                              void* d_out, int out_size, void* d_ws, size_t ws_size,
                              hipStream_t stream) {
  (void)in_sizes; (void)n_in; (void)out_size; (void)ws_size;
  const float* tgt    = (const float*)d_in[0];
  const float* memory = (const float*)d_in[1];
  const float* Wq = (const float*)d_in[2];
  const float* bq = (const float*)d_in[3];
  const float* Wk = (const float*)d_in[4];
  const float* bk = (const float*)d_in[5];
  const float* Wv = (const float*)d_in[6];
  const float* bv = (const float*)d_in[7];
  const float* Wo = (const float*)d_in[8];
  const float* bo = (const float*)d_in[9];
  float* out = (float*)d_out;

  char* ws = (char*)d_ws;
  const size_t WBYTES = (size_t)DMODEL * DMODEL * 2;              // 2 MB each
  const size_t XBYTES = (size_t)T_SEQ * BATCH * DMODEL * 2;       // 32 MB each
  const size_t PBYTES = (size_t)BATCH * NHEAD * T_SEQ * HDIM * 2; // 32 MB each
  __bf16* WqBf = (__bf16*)(ws + 0 * WBYTES);
  __bf16* WkBf = (__bf16*)(ws + 1 * WBYTES);
  __bf16* WvBf = (__bf16*)(ws + 2 * WBYTES);
  __bf16* WoBf = (__bf16*)(ws + 3 * WBYTES);
  size_t off = 4 * WBYTES;
  __bf16* tgtBf = (__bf16*)(ws + off); off += XBYTES;             // [t,b,D]
  __bf16* memBf = (__bf16*)(ws + off); off += XBYTES;             // [t,b,D]
  __bf16* qp    = (__bf16*)(ws + off); off += PBYTES;             // [b,h,t,d]
  __bf16* kpT   = (__bf16*)(ws + off); off += PBYTES;             // [b,h,d,T]
  __bf16* vT    = (__bf16*)(ws + off); off += PBYTES;             // [b,h,d,T]
  __bf16* ctxT  = (__bf16*)(ws + off);
  off += (size_t)BATCH * NHEAD * HDIM * HDIM * 2;                 // [b,h,e,d]
  __bf16* attn  = (__bf16*)(ws + off);                            // [b,t,D]

  const int NW8 = DMODEL * DMODEL / 8;
  const int NX8 = T_SEQ * BATCH * DMODEL / 8;
  convert_bf16x8_kernel<<<1024, 256, 0, stream>>>(Wq, WqBf, NW8);
  convert_bf16x8_kernel<<<1024, 256, 0, stream>>>(Wk, WkBf, NW8);
  convert_bf16x8_kernel<<<1024, 256, 0, stream>>>(Wv, WvBf, NW8);
  convert_bf16x8_kernel<<<1024, 256, 0, stream>>>(Wo, WoBf, NW8);
  convert_bf16x8_kernel<<<2048, 256, 0, stream>>>(tgt, tgtBf, NX8);
  convert_bf16x8_kernel<<<2048, 256, 0, stream>>>(memory, memBf, NX8);

  dim3 gq(T_SEQ / 128, DMODEL / 64, BATCH);
  qkv_proj_kernel<<<gq, 256, 0, stream>>>(tgtBf, WqBf, bq, qp,  0);
  qkv_proj_kernel<<<gq, 256, 0, stream>>>(memBf, WkBf, bk, kpT, 1);
  qkv_proj_kernel<<<gq, 256, 0, stream>>>(memBf, WvBf, bv, vT,  2);

  context_kernel<<<BATCH * NHEAD, 256, 0, stream>>>(kpT, vT, ctxT);

  dim3 ga(T_SEQ / 128, BATCH * NHEAD);
  attn_kernel<<<ga, 256, 0, stream>>>(qp, ctxT, attn);

  out_proj_kernel<<<gq, 256, 0, stream>>>(attn, WoBf, bo, out);
}
